// Model_75316546502984
// MI455X (gfx1250) — matrix operand order repaired and checked by reading, not compiled, not run on hardware
//
#include <hip/hip_runtime.h>
#include <math.h>
#include <stdint.h>

typedef __attribute__((ext_vector_type(16))) _Float16 v16h;
typedef __attribute__((ext_vector_type(8)))  _Float16 v8h;
typedef __attribute__((ext_vector_type(8)))  float    v8f;

#define B_     64
#define S_     360
#define DIN    219
#define DM     512
#define HIDN   1024
#define G4     4096
#define KXPAD  224
#define KLIN   1536
#define NLPAD  224
#define NJOINT 24
#define MROWS  (B_*S_)

__constant__ int c_parents[NJOINT] = {-1,0,0,0,1,2,3,4,5,6,7,8,9,9,9,12,13,14,16,17,18,19,20,21};

__device__ __forceinline__ float sigm(float x){ return 1.0f/(1.0f+expf(-x)); }

// ---- convert f32 weights -> f16 in WMMA B-fragment order ----
__global__ void cvt_swz_kernel(const float* __restrict__ src, _Float16* __restrict__ dst,
                               int K, int N, int NT, int total) {
  int idx = blockIdx.x * blockDim.x + threadIdx.x;
  if (idx >= total) return;
  int e    = idx & 15;
  int lane = (idx >> 4) & 31;
  int nt   = (idx >> 9) % NT;
  int kt   = (idx >> 9) / NT;
  int k = kt * 32 + ((e >> 3) << 4) + ((lane >> 4) << 3) + (e & 7);
  int n = nt * 16 + (lane & 15);
  dst[idx] = (k < K && n < N) ? (_Float16)src[(size_t)k * N + n] : (_Float16)0.f;
}

// ---- simple linear f32 -> f16 convert ----
__global__ void cvt_lin_kernel(const float* __restrict__ src, _Float16* __restrict__ dst, int n) {
  int i = blockIdx.x * blockDim.x + threadIdx.x;
  if (i < n) dst[i] = (_Float16)src[i];
}

// ---------------- encoder: relu(src @ W + b + pos_emb[pos]) -> f16 -----------
__global__ void enc_kernel(const float* __restrict__ src, const v16h* __restrict__ Wf,
                           const float* __restrict__ bias, const float* __restrict__ pos_emb,
                           const int* __restrict__ src_pos, _Float16* __restrict__ encO) {
  int wave = (blockIdx.x * blockDim.x + threadIdx.x) >> 5;
  int lane = threadIdx.x & 31;
  int mt = wave >> 5, nt = wave & 31;
  int m0 = mt * 16, n0 = nt * 16;
  int lrow = lane & 15, lhi = lane >> 4;
  v8f acc = {};
  for (int k0 = 0; k0 < KXPAD; k0 += 32) {
    union { v16h v; _Float16 x[16]; } a, b;
    int m = m0 + lrow;
    #pragma unroll
    for (int e = 0; e < 16; ++e) {
      int k = k0 + ((e >> 3) << 4) + (lhi << 3) + (e & 7);
      a.x[e] = (k < DIN) ? (_Float16)src[(size_t)m * DIN + k] : (_Float16)0.f;
    }
    b.v = Wf[((size_t)(k0 >> 5) * 32 + nt) * 32 + lane];
    acc = __builtin_amdgcn_wmma_f32_16x16x32_f16(false, a.v, false, b.v, (short)0, acc, false, false);
  }
  int n = n0 + lrow;
  float bn = bias[n];
  #pragma unroll
  for (int r = 0; r < 8; ++r) {
    int m = m0 + (lhi << 3) + r;
    int pos = src_pos[m];
    float v = acc[r] + bn + pos_emb[(size_t)pos * DM + n];
    v = v > 0.f ? v : 0.f;
    encO[(size_t)m * DM + n] = (_Float16)v;
  }
}

// ---------- fused LSTM step: gates GEMM (x@Wx + h@Wh + b) + cell update ------
// 2 m-groups (32 rows) * 64 j-tiles = 128 waves -> 16 blocks x 256 threads.
// h rows are staged into LDS with GLOBAL_LOAD_ASYNC_TO_LDS_B128 (contiguous 64 KB).
__global__ void lstm_step_kernel(const float* __restrict__ tgt, const float* __restrict__ prev_out,
                                 const float* __restrict__ dec0,
                                 const _Float16* __restrict__ h_in, _Float16* __restrict__ h_out,
                                 float* __restrict__ c,
                                 const v16h* __restrict__ Wxf, const v16h* __restrict__ Whf,
                                 const float* __restrict__ b_lstm, const int* __restrict__ epoch,
                                 int t) {
  __shared__ _Float16 sx[32 * KXPAD];   // 14 KB (f32->f16 convert path)
  __shared__ _Float16 sh[32 * HIDN];    // 64 KB (raw async copy path)

  int wave = (blockIdx.x * blockDim.x + threadIdx.x) >> 5;
  int lane = threadIdx.x & 31;
  int mg = wave >> 6, jt = wave & 63;    // m-group (0/1), j-tile
  int mbase = mg * 32;
  int j0 = jt * 16;
  int lrow = lane & 15, lhi = lane >> 4;

  int n_pred = (int)((double)epoch[0] * 0.01);
  int period = n_pred + 10;
  bool teacher = (t % period) >= n_pred;

  const float* xbase; long xs;
  if (teacher)      { xbase = tgt + (size_t)t * DIN;              xs = (long)S_ * DIN; }
  else if (t == 0)  { xbase = dec0;                               xs = DIN; }
  else              { xbase = prev_out + (size_t)(t - 1) * DIN;   xs = (long)S_ * DIN; }

  // ---- async copy of 32 h rows (contiguous 64 KB, f16) into LDS ----
  {
    const _Float16* hsrc = h_in + (size_t)mbase * HIDN;
    uint32_t lds0 = (uint32_t)(uintptr_t)(&sh[0]);
    for (int ch = threadIdx.x; ch < (32 * HIDN) / 8; ch += 256) {
      uint32_t loff = lds0 + (uint32_t)ch * 16u;
      uint64_t ga = (uint64_t)(uintptr_t)(hsrc + (size_t)ch * 8);
      asm volatile("global_load_async_to_lds_b128 %0, %1, off"
                   :: "v"(loff), "v"(ga) : "memory");
    }
  }
  // ---- stage x rows (f32 -> f16 with K padding) ----
  for (int i = threadIdx.x; i < 32 * KXPAD; i += 256) {
    int r = i / KXPAD, k = i - r * KXPAD;
    sx[i] = (k < DIN) ? (_Float16)xbase[(size_t)(mbase + r) * xs + k] : (_Float16)0.f;
  }
  asm volatile("s_wait_asynccnt 0x0" ::: "memory");
  __syncthreads();

  v8f acc[2][4] = {};

  // ---- x @ Wx (K padded to 224) ----
  for (int k0 = 0; k0 < KXPAD; k0 += 32) {
    union { v16h v; v8h h2[2]; } a[2];
    #pragma unroll
    for (int s = 0; s < 2; ++s) {
      const _Float16* arow = &sx[(s * 16 + lrow) * KXPAD];
      a[s].h2[0] = *(const v8h*)&arow[k0 + (lhi << 3)];
      a[s].h2[1] = *(const v8h*)&arow[k0 + 16 + (lhi << 3)];
    }
    #pragma unroll
    for (int g = 0; g < 4; ++g) {
      v16h bv = Wxf[((size_t)(k0 >> 5) * 256 + (g * 64 + jt)) * 32 + lane];
      acc[0][g] = __builtin_amdgcn_wmma_f32_16x16x32_f16(false, a[0].v, false, bv, (short)0, acc[0][g], false, false);
      acc[1][g] = __builtin_amdgcn_wmma_f32_16x16x32_f16(false, a[1].v, false, bv, (short)0, acc[1][g], false, false);
    }
  }

  // ---- h @ Wh (K = 1024) ----
  for (int k0 = 0; k0 < HIDN; k0 += 32) {
    if (k0 + 32 < HIDN)
      __builtin_prefetch(&Whf[((size_t)((k0 + 32) >> 5) * 256 + jt) * 32 + lane], 0, 3);
    union { v16h v; v8h h2[2]; } a[2];
    #pragma unroll
    for (int s = 0; s < 2; ++s) {
      const _Float16* arow = &sh[(s * 16 + lrow) * HIDN];
      a[s].h2[0] = *(const v8h*)&arow[k0 + (lhi << 3)];
      a[s].h2[1] = *(const v8h*)&arow[k0 + 16 + (lhi << 3)];
    }
    #pragma unroll
    for (int g = 0; g < 4; ++g) {
      v16h bv = Whf[((size_t)(k0 >> 5) * 256 + (g * 64 + jt)) * 32 + lane];
      acc[0][g] = __builtin_amdgcn_wmma_f32_16x16x32_f16(false, a[0].v, false, bv, (short)0, acc[0][g], false, false);
      acc[1][g] = __builtin_amdgcn_wmma_f32_16x16x32_f16(false, a[1].v, false, bv, (short)0, acc[1][g], false, false);
    }
  }

  // ---- elementwise cell update ----
  int j = j0 + lrow;
  float bi = b_lstm[j], bf = b_lstm[HIDN + j], bg = b_lstm[2 * HIDN + j], bo = b_lstm[3 * HIDN + j];
  #pragma unroll
  for (int s = 0; s < 2; ++s) {
    #pragma unroll
    for (int r = 0; r < 8; ++r) {
      int m = mbase + s * 16 + (lhi << 3) + r;
      float iv = acc[s][0][r] + bi;
      float fv = acc[s][1][r] + bf;
      float gv = acc[s][2][r] + bg;
      float ov = acc[s][3][r] + bo;
      float cm = c[(size_t)m * HIDN + j];
      float cn = sigm(fv) * cm + sigm(iv) * tanhf(gv);
      float hn = sigm(ov) * tanhf(cn);
      c[(size_t)m * HIDN + j] = cn;
      h_out[(size_t)m * HIDN + j] = (_Float16)hn;
    }
  }
}

// ---------- output projection: concat(h, e_t) @ W_lin + b_lin -> out[:,t,:] ---
__global__ void proj_kernel(const _Float16* __restrict__ h, const _Float16* __restrict__ encO,
                            const v16h* __restrict__ Wlf, const float* __restrict__ b_lin,
                            float* __restrict__ out, int t) {
  int wave = (blockIdx.x * blockDim.x + threadIdx.x) >> 5;
  int lane = threadIdx.x & 31;
  int mt = wave / 14, nt = wave % 14;
  int m0 = mt * 16, n0 = nt * 16;
  int lrow = lane & 15, lhi = lane >> 4;
  v8f acc = {};
  for (int k0 = 0; k0 < KLIN; k0 += 32) {
    union { v16h v; v8h h2[2]; } a;
    int m = m0 + lrow;
    const _Float16* arow = (k0 < HIDN)
        ? (h + (size_t)m * HIDN + k0)
        : (encO + (size_t)m * S_ * DM + (size_t)t * DM + (k0 - HIDN));
    a.h2[0] = *(const v8h*)&arow[(lhi << 3)];
    a.h2[1] = *(const v8h*)&arow[16 + (lhi << 3)];
    v16h bv = Wlf[((size_t)(k0 >> 5) * 14 + nt) * 32 + lane];
    acc = __builtin_amdgcn_wmma_f32_16x16x32_f16(false, a.v, false, bv, (short)0, acc, false, false);
  }
  int n = n0 + lrow;
  if (n < DIN) {
    float bn = b_lin[n];
    #pragma unroll
    for (int r = 0; r < 8; ++r) {
      int m = m0 + (lhi << 3) + r;
      out[(size_t)m * S_ * DIN + (size_t)t * DIN + n] = acc[r] + bn;
    }
  }
}

// ------- FK joints: pose[...,:216] -> rotmat -> aa -> rodrigues -> FK --------
__global__ void joints_kernel(const float* __restrict__ pose, const float* __restrict__ J_rest,
                              float* __restrict__ joints) {
  int n = blockIdx.x * blockDim.x + threadIdx.x;
  if (n >= MROWS) return;
  const float* p = pose + (size_t)n * DIN;
  float Rg[NJOINT][9];
  float tg[NJOINT * 3];
  for (int j = 0; j < NJOINT; ++j) {
    float R[9];
    #pragma unroll
    for (int q = 0; q < 9; ++q) R[q] = p[j * 9 + q];
    float tr = R[0] + R[4] + R[8];
    float cs = 0.5f * (tr - 1.0f);
    cs = fminf(fmaxf(cs, -1.0f + 1e-6f), 1.0f - 1e-6f);
    float ang = acosf(cs);
    float sc = ang / (2.0f * sinf(ang) + 1e-8f);
    float a0 = (R[7] - R[5]) * sc, a1 = (R[2] - R[6]) * sc, a2 = (R[3] - R[1]) * sc;
    float an = sqrtf(a0 * a0 + a1 * a1 + a2 * a2 + 1e-8f);
    float x = a0 / an, y = a1 / an, z = a2 / an;
    float sn = sinf(an), cn1 = 1.0f - cosf(an);
    float Rr[9];
    Rr[0] = 1.0f - cn1 * (z * z + y * y);
    Rr[1] = -sn * z + cn1 * (x * y);
    Rr[2] =  sn * y + cn1 * (x * z);
    Rr[3] =  sn * z + cn1 * (x * y);
    Rr[4] = 1.0f - cn1 * (x * x + z * z);
    Rr[5] = -sn * x + cn1 * (y * z);
    Rr[6] = -sn * y + cn1 * (x * z);
    Rr[7] =  sn * x + cn1 * (y * z);
    Rr[8] = 1.0f - cn1 * (x * x + y * y);
    float tx, ty, tz;
    if (j == 0) {
      #pragma unroll
      for (int q = 0; q < 9; ++q) Rg[0][q] = Rr[q];
      tx = J_rest[0]; ty = J_rest[1]; tz = J_rest[2];
    } else {
      int par = c_parents[j];
      float* Rp = Rg[par];
      float* Rj = Rg[j];
      for (int a = 0; a < 3; ++a)
        for (int bq = 0; bq < 3; ++bq)
          Rj[a * 3 + bq] = Rp[a * 3] * Rr[bq] + Rp[a * 3 + 1] * Rr[3 + bq] + Rp[a * 3 + 2] * Rr[6 + bq];
      float rx = J_rest[j * 3]     - J_rest[par * 3];
      float ry = J_rest[j * 3 + 1] - J_rest[par * 3 + 1];
      float rz = J_rest[j * 3 + 2] - J_rest[par * 3 + 2];
      tx = tg[par * 3]     + Rp[0] * rx + Rp[1] * ry + Rp[2] * rz;
      ty = tg[par * 3 + 1] + Rp[3] * rx + Rp[4] * ry + Rp[5] * rz;
      tz = tg[par * 3 + 2] + Rp[6] * rx + Rp[7] * ry + Rp[8] * rz;
    }
    tg[j * 3] = tx; tg[j * 3 + 1] = ty; tg[j * 3 + 2] = tz;
    joints[(size_t)n * (NJOINT * 3) + j * 3]     = tx;
    joints[(size_t)n * (NJOINT * 3) + j * 3 + 1] = ty;
    joints[(size_t)n * (NJOINT * 3) + j * 3 + 2] = tz;
  }
}

extern "C" void kernel_launch(void* const* d_in, const int* in_sizes, int n_in,
                              void* d_out, int out_size, void* d_ws, size_t ws_size,
                              hipStream_t stream) {
  (void)in_sizes; (void)n_in; (void)out_size; (void)ws_size;
  const float* src_seq = (const float*)d_in[0];
  const float* tgt_seq = (const float*)d_in[1];
  const float* gt      = (const float*)d_in[2];
  const float* vec_h   = (const float*)d_in[3];
  const float* vec_c   = (const float*)d_in[4];
  const float* dec0    = (const float*)d_in[5];
  const int*   src_pos = (const int*)d_in[7];
  const int*   epoch   = (const int*)d_in[8];
  const float* enc_W   = (const float*)d_in[9];
  const float* enc_b   = (const float*)d_in[10];
  const float* pos_emb = (const float*)d_in[11];
  const float* Wx      = (const float*)d_in[12];
  const float* Wh      = (const float*)d_in[13];
  const float* b_lstm  = (const float*)d_in[14];
  const float* W_lin   = (const float*)d_in[15];
  const float* b_lin   = (const float*)d_in[16];
  const float* J_rest  = (const float*)d_in[17];
  float* out = (float*)d_out;

  // workspace layout (bytes, all 32B aligned)
  char* ws = (char*)d_ws;
  size_t off = 0;
  _Float16* encWf = (_Float16*)(ws + off); off += (size_t)KXPAD * DM * 2;
  _Float16* Wxf   = (_Float16*)(ws + off); off += (size_t)KXPAD * G4 * 2;
  _Float16* Whf   = (_Float16*)(ws + off); off += (size_t)HIDN * G4 * 2;
  _Float16* Wlf   = (_Float16*)(ws + off); off += (size_t)KLIN * NLPAD * 2;
  _Float16* encO  = (_Float16*)(ws + off); off += (size_t)MROWS * DM * 2;
  _Float16* hf0   = (_Float16*)(ws + off); off += (size_t)B_ * HIDN * 2;
  _Float16* hf1   = (_Float16*)(ws + off); off += (size_t)B_ * HIDN * 2;
  float*    c     = (float*)(ws + off);    off += (size_t)B_ * HIDN * 4;

  hipMemcpyAsync(c, vec_c, (size_t)B_ * HIDN * 4, hipMemcpyDeviceToDevice, stream);
  cvt_lin_kernel<<<(B_ * HIDN + 255) / 256, 256, 0, stream>>>(vec_h, hf0, B_ * HIDN);

  // weight conversion + fragment swizzle
  {
    int cnt;
    cnt = KXPAD * DM;     // KT=7,  NT=32
    cvt_swz_kernel<<<(cnt + 255) / 256, 256, 0, stream>>>(enc_W, encWf, DIN, DM, 32, cnt);
    cnt = KXPAD * G4;     // KT=7,  NT=256
    cvt_swz_kernel<<<(cnt + 255) / 256, 256, 0, stream>>>(Wx, Wxf, DIN, G4, 256, cnt);
    cnt = HIDN * G4;      // KT=32, NT=256
    cvt_swz_kernel<<<(cnt + 255) / 256, 256, 0, stream>>>(Wh, Whf, HIDN, G4, 256, cnt);
    cnt = KLIN * NLPAD;   // KT=48, NT=14
    cvt_swz_kernel<<<(cnt + 255) / 256, 256, 0, stream>>>(W_lin, Wlf, KLIN, DIN, 14, cnt);
  }

  // encoder
  enc_kernel<<<5760, 256, 0, stream>>>(src_seq, (const v16h*)encWf, enc_b, pos_emb, src_pos, encO);

  // sequential decoder: 360 steps, 2 kernels/step
  for (int t = 0; t < S_; ++t) {
    const _Float16* h_in = (t & 1) ? hf1 : hf0;
    _Float16*       h_ot = (t & 1) ? hf0 : hf1;
    lstm_step_kernel<<<16, 256, 0, stream>>>(tgt_seq, out, dec0, h_in, h_ot, c,
                                             (const v16h*)Wxf, (const v16h*)Whf, b_lstm, epoch, t);
    proj_kernel<<<7, 256, 0, stream>>>(h_ot, encO, (const v16h*)Wlf, b_lin, out, t);
  }

  // joints for prediction and ground truth
  float* pred_j = out + (size_t)MROWS * DIN;
  float* targ_j = pred_j + (size_t)MROWS * NJOINT * 3;
  joints_kernel<<<(MROWS + 255) / 256, 256, 0, stream>>>(out, J_rest, pred_j);
  joints_kernel<<<(MROWS + 255) / 256, 256, 0, stream>>>(gt,  J_rest, targ_j);
}
